// ENC_DECCell_47399259079246
// MI455X (gfx1250) — matrix operand order repaired and checked by reading, not compiled, not run on hardware
//
#include <hip/hip_runtime.h>

// ---------------- problem constants ----------------
#define N_ROWS 4096
#define SEQ_L  32
#define HID    256
#define CDIM   256
#define DEC_T  16
#define GATE_E 768     // 3*HID
#define GATE_D 1536    // 6*HID
#define S2DIM  512     // 2*HID
#define ROWS_PER_BLK 16
#define NTHREADS 256

typedef __attribute__((ext_vector_type(16))) __bf16 bf16x16;
typedef __attribute__((ext_vector_type(8)))  float  f32x8;

// bf16 weight workspace offsets (element counts)
#define OFF_WIHE 0
#define OFF_WHHE (OFF_WIHE + GATE_E*HID)       // 196608
#define OFF_WIHD (OFF_WHHE + GATE_E*HID)       // 393216
#define OFF_WHHD (OFF_WIHD + GATE_D*HID)       // 786432
#define OFF_WRED (OFF_WHHD + GATE_D*S2DIM)     // 1572864
#define OFF_WOUT (OFF_WRED + HID*S2DIM)        // 1703936
#define WS_ELEMS (OFF_WOUT + CDIM*HID)         // 1769472

static __device__ __forceinline__ unsigned short f2bf(float f) {
  union { float f; unsigned u; } v; v.f = f;
  unsigned r = (v.u + 0x7FFFu + ((v.u >> 16) & 1u)) >> 16;
  return (unsigned short)r;
}
static __device__ __forceinline__ float bf2f(unsigned short b) {
  union { unsigned u; float f; } v; v.u = ((unsigned)b) << 16;
  return v.f;
}
static __device__ __forceinline__ float sigm(float x) {
  return 1.0f / (1.0f + __expf(-x));
}

// A operand (16x32 bf16 tile): lane L holds row M=L&15;
// halves 0..7 = K kb..kb+7, halves 8..15 = K kb+16..kb+23, kb = (L>=16)*8.
static __device__ __forceinline__ bf16x16 load_a(const unsigned short* base,
                                                 int ld, int k0, int lane) {
  const unsigned short* p = base + (lane & 15) * ld + k0 + ((lane >> 4) << 3);
  union { uint4 q[2]; bf16x16 v; } u;
  u.q[0] = *(const uint4*)(p);
  u.q[1] = *(const uint4*)(p + 16);
  return u.v;
}
// B operand (32x16 bf16 tile), B[k][n] = W[n][k]: lane L holds col N=L&15,
// halves 0..7 = K kb..kb+7, halves 8..15 = K kb+16..kb+23, kb = k0 + (L>=16)*8 (W row-major, row = out col).
static __device__ __forceinline__ bf16x16 load_b(const unsigned short* w,
                                                 int col, int ld, int k0, int lane) {
  const unsigned short* p = w + col * ld + k0 + ((lane >> 4) << 3);
  union { uint4 q[2]; bf16x16 v; } u;
  u.q[0] = *(const uint4*)(p);
  u.q[1] = *(const uint4*)(p + 16);
  return u.v;
}

// One 16x16 output tile of  A(16xK) * W^T(Kx16) + bias[col], fp32 accum via WMMA.
template <int K>
static __device__ __forceinline__ f32x8 mm_tile(const unsigned short* A, int ldA,
                                                const unsigned short* W, int ldW,
                                                int colbase,
                                                const float* __restrict__ bias,
                                                int lane) {
  int col = colbase + (lane & 15);
  float b0 = bias[col];
  f32x8 acc;
#pragma unroll
  for (int i = 0; i < 8; ++i) acc[i] = b0;
#pragma unroll
  for (int k0 = 0; k0 < K; k0 += 32) {
    bf16x16 a = load_a(A, ldA, k0, lane);
    bf16x16 b = load_b(W, col, ldW, k0, lane);
    acc = __builtin_amdgcn_wmma_f32_16x16x32_bf16(false, a, false, b,
                                                  (short)0, acc, false, false);
  }
  return acc;
}

// ---------------- fp32 -> bf16 weight conversion ----------------
__global__ void cvt_weights(const float* __restrict__ s0, const float* __restrict__ s1,
                            const float* __restrict__ s2, const float* __restrict__ s3,
                            const float* __restrict__ s4, const float* __restrict__ s5,
                            unsigned short* __restrict__ d) {
  const int n0 = GATE_E*HID, n1 = GATE_E*HID, n2 = GATE_D*HID,
            n3 = GATE_D*S2DIM, n4 = HID*S2DIM, n5 = CDIM*HID;
  const int total = n0 + n1 + n2 + n3 + n4 + n5;
  for (int i = blockIdx.x * blockDim.x + threadIdx.x; i < total;
       i += gridDim.x * blockDim.x) {
    float v; int j = i;
    if (j < n0) v = s0[j];
    else { j -= n0; if (j < n1) v = s1[j];
    else { j -= n1; if (j < n2) v = s2[j];
    else { j -= n2; if (j < n3) v = s3[j];
    else { j -= n3; if (j < n4) v = s4[j];
    else v = s5[j - n4]; } } } }
    d[i] = f2bf(v);
  }
}

// ---------------- dynamic LDS layout ----------------
#define SM_ARG   0                              // 16 * u64      = 128
#define SM_ALIVE 128                            // 16 * int      = 64
#define SM_G     192                            // 16*1536 f32   = 98304
#define SM_S2    (SM_G  + ROWS_PER_BLK*GATE_D*4)      // 16*512 bf16 = 16384
#define SM_X     (SM_S2 + ROWS_PER_BLK*S2DIM*2)       // 16*256 bf16 = 8192
#define SM_N2    (SM_X  + ROWS_PER_BLK*HID*2)         // 16*512 bf16 = 16384
#define SM_NH    (SM_N2 + ROWS_PER_BLK*S2DIM*2)       // 16*256 bf16 = 8192
#define SMEM_BYTES (SM_NH + ROWS_PER_BLK*HID*2)       // 147648

// ---------------- persistent fused encoder+decoder ----------------
__global__ __launch_bounds__(NTHREADS, 1)
void tree_gru(const float* __restrict__ anc,  const float* __restrict__ encg,
              const float* __restrict__ bihe, const float* __restrict__ bhhe,
              const float* __restrict__ bihd, const float* __restrict__ bhhd,
              const float* __restrict__ bred, const float* __restrict__ bout,
              const unsigned short* __restrict__ Wihe,
              const unsigned short* __restrict__ Whhe,
              const unsigned short* __restrict__ Wihd,
              const unsigned short* __restrict__ Whhd,
              const unsigned short* __restrict__ Wred,
              const unsigned short* __restrict__ Wout,
              float* __restrict__ out) {
  extern __shared__ char smem[];
  unsigned long long* sArg = (unsigned long long*)(smem + SM_ARG);
  int*            sAlive = (int*)(smem + SM_ALIVE);
  float*          sG     = (float*)(smem + SM_G);
  unsigned short* sS2    = (unsigned short*)(smem + SM_S2);  // [hid | enc] bf16
  unsigned short* sX     = (unsigned short*)(smem + SM_X);   // step input bf16
  unsigned short* sN2    = (unsigned short*)(smem + SM_N2);  // decoder new2 bf16
  unsigned short* sNH    = (unsigned short*)(smem + SM_NH);  // decoder new_h bf16

  const int tid  = threadIdx.x;
  const int wave = tid >> 5;
  const int lane = tid & 31;
  const int hi8  = (lane >> 4) << 3;   // row offset for C/D layout
  const int ncol = lane & 15;
  const int row0 = blockIdx.x * ROWS_PER_BLK;

  // h = 0 (and clear enc half; loaded later)
  for (int i = tid; i < ROWS_PER_BLK * S2DIM; i += NTHREADS) sS2[i] = 0;
  __syncthreads();

  // ================= encoder: 32 GRU steps =================
  for (int t = 0; t < SEQ_L; ++t) {
    for (int i = tid; i < ROWS_PER_BLK * HID; i += NTHREADS) {
      int m = i >> 8, c = i & 255;
      sX[i] = f2bf(anc[((size_t)(row0 + m) * SEQ_L + t) * HID + c]);
    }
    __syncthreads();
    // gh = h @ W_hh_e^T + b_hh_e  -> sG (48 tiles of 16x16)
#pragma unroll 1
    for (int tt = wave; tt < 48; tt += 8) {
      int cb = tt * 16;
      f32x8 acc = mm_tile<HID>(sS2, S2DIM, Whhe, HID, cb, bhhe, lane);
#pragma unroll
      for (int i = 0; i < 8; ++i) sG[(i + hi8) * GATE_D + cb + ncol] = acc[i];
    }
    __syncthreads();
    // gi(r,z) = x @ W_ih_e^T + b_ih_e, accumulate into sG (tiles 0..31)
#pragma unroll 1
    for (int tt = wave; tt < 32; tt += 8) {
      int cb = tt * 16;
      f32x8 acc = mm_tile<HID>(sX, HID, Wihe, HID, cb, bihe, lane);
#pragma unroll
      for (int i = 0; i < 8; ++i) sG[(i + hi8) * GATE_D + cb + ncol] += acc[i];
    }
    __syncthreads();
    // gi(n) tiles 32..47: compute and immediately apply gate elementwise
#pragma unroll 1
    for (int tt = 32 + wave; tt < 48; tt += 8) {
      int cb = tt * 16;
      f32x8 acc = mm_tile<HID>(sX, HID, Wihe, HID, cb, bihe, lane);
      int j = (tt - 32) * 16 + ncol;  // hidden col 0..255
#pragma unroll
      for (int i = 0; i < 8; ++i) {
        int m = i + hi8;
        float r  = sigm(sG[m * GATE_D + j]);
        float z  = sigm(sG[m * GATE_D + HID + j]);
        float hn = sG[m * GATE_D + 2 * HID + j];
        float ho = bf2f(sS2[m * S2DIM + j]);
        float hv = (1.f - z) * tanhf(acc[i] + r * hn) + z * ho;
        sS2[m * S2DIM + j] = f2bf(hv);
      }
    }
    __syncthreads();
  }

  // ================= label = h @ W_out^T + b_out ; x = relu(label) =========
  for (int i = tid; i < ROWS_PER_BLK * HID; i += NTHREADS) {
    int m = i >> 8, c = i & 255;
    sS2[m * S2DIM + HID + c] = f2bf(encg[(size_t)(row0 + m) * HID + c]);
  }
  if (tid < 16) { sAlive[tid] = 1; sArg[tid] = 0ull; }
  __syncthreads();
#pragma unroll 1
  for (int tt = wave; tt < 16; tt += 8) {
    int cb = tt * 16;
    f32x8 acc = mm_tile<HID>(sS2, S2DIM, Wout, HID, cb, bout, lane);
#pragma unroll
    for (int i = 0; i < 8; ++i)
      sX[(i + hi8) * HID + cb + ncol] = f2bf(fmaxf(acc[i], 0.f));
  }
  __syncthreads();

  // ================= decoder: 16 steps =================
  for (int t = 0; t < DEC_T; ++t) {
    // gh = state2 @ W_hh_d^T + b_hh_d  (96 tiles, K=512)
#pragma unroll 1
    for (int tt = wave; tt < 96; tt += 8) {
      int cb = tt * 16;
      f32x8 acc = mm_tile<S2DIM>(sS2, S2DIM, Whhd, S2DIM, cb, bhhd, lane);
#pragma unroll
      for (int i = 0; i < 8; ++i) sG[(i + hi8) * GATE_D + cb + ncol] = acc[i];
    }
    __syncthreads();
    // gi(r,z) = x @ W_ih_d^T + b_ih_d (tiles 0..63, K=256), accumulate
#pragma unroll 1
    for (int tt = wave; tt < 64; tt += 8) {
      int cb = tt * 16;
      f32x8 acc = mm_tile<HID>(sX, HID, Wihd, HID, cb, bihd, lane);
#pragma unroll
      for (int i = 0; i < 8; ++i) sG[(i + hi8) * GATE_D + cb + ncol] += acc[i];
    }
    __syncthreads();
    // gi(n) tiles 64..95: compute, apply GRU gates, write new2 -> sN2
#pragma unroll 1
    for (int tt = 64 + wave; tt < 96; tt += 8) {
      int cb = tt * 16;
      f32x8 acc = mm_tile<HID>(sX, HID, Wihd, HID, cb, bihd, lane);
      int j = (tt - 64) * 16 + ncol;  // state2 col 0..511
#pragma unroll
      for (int i = 0; i < 8; ++i) {
        int m = i + hi8;
        float r  = sigm(sG[m * GATE_D + j]);
        float z  = sigm(sG[m * GATE_D + S2DIM + j]);
        float hn = sG[m * GATE_D + 2 * S2DIM + j];
        float s2 = bf2f(sS2[m * S2DIM + j]);
        float v  = (1.f - z) * tanhf(acc[i] + r * hn) + z * s2;
        sN2[m * S2DIM + j] = f2bf(v);
      }
    }
    __syncthreads();
    // new_h = new2 @ W_red^T + b_red ; masked hid update
#pragma unroll 1
    for (int tt = wave; tt < 16; tt += 8) {
      int cb = tt * 16;
      f32x8 acc = mm_tile<S2DIM>(sN2, S2DIM, Wred, S2DIM, cb, bred, lane);
#pragma unroll
      for (int i = 0; i < 8; ++i) {
        int m = i + hi8;
        unsigned short hb = f2bf(acc[i]);
        sNH[m * HID + cb + ncol] = hb;
        if (sAlive[m]) sS2[m * S2DIM + cb + ncol] = hb;  // hid = where(alive,...)
      }
    }
    __syncthreads();
    // new_lab = new_h @ W_out^T + b_out ; output, relu-x, argmax
#pragma unroll 1
    for (int tt = wave; tt < 16; tt += 8) {
      int cb = tt * 16;
      f32x8 acc = mm_tile<HID>(sNH, HID, Wout, HID, cb, bout, lane);
#pragma unroll
      for (int i = 0; i < 8; ++i) {
        int m = i + hi8;
        int c = cb + ncol;
        float v = acc[i];
        int alive = sAlive[m];
        out[((size_t)(row0 + m) * DEC_T + t) * CDIM + c] = alive ? v : 0.f;
        if (alive) sX[m * HID + c] = f2bf(fmaxf(v, 0.f));
        unsigned ub = __float_as_uint(v);
        ub = (ub & 0x80000000u) ? ~ub : (ub | 0x80000000u);  // order-preserving
        unsigned long long pk =
            ((unsigned long long)ub << 32) | (unsigned)(255 - c);  // first-max ties
        atomicMax(&sArg[m], pk);
      }
    }
    __syncthreads();
    if (tid < 16) {
      int am = 255 - (int)(sArg[tid] & 0xFFFFFFFFull);
      if (am == 0) sAlive[tid] = 0;  // EOS == 0
      sArg[tid] = 0ull;
    }
    __syncthreads();
  }
}

// ---------------- launch ----------------
extern "C" void kernel_launch(void* const* d_in, const int* in_sizes, int n_in,
                              void* d_out, int out_size, void* d_ws, size_t ws_size,
                              hipStream_t stream) {
  const float* anc  = (const float*)d_in[0];
  const float* enc  = (const float*)d_in[1];
  const float* Wihe = (const float*)d_in[2];
  const float* Whhe = (const float*)d_in[3];
  const float* bihe = (const float*)d_in[4];
  const float* bhhe = (const float*)d_in[5];
  const float* Wihd = (const float*)d_in[6];
  const float* Whhd = (const float*)d_in[7];
  const float* bihd = (const float*)d_in[8];
  const float* bhhd = (const float*)d_in[9];
  const float* Wred = (const float*)d_in[10];
  const float* bred = (const float*)d_in[11];
  const float* Wout = (const float*)d_in[12];
  const float* bout = (const float*)d_in[13];
  unsigned short* ws = (unsigned short*)d_ws;

  cvt_weights<<<1024, 256, 0, stream>>>(Wihe, Whhe, Wihd, Whhd, Wred, Wout, ws);

  tree_gru<<<N_ROWS / ROWS_PER_BLK, NTHREADS, SMEM_BYTES, stream>>>(
      anc, enc, bihe, bhhe, bihd, bhhd, bred, bout,
      ws + OFF_WIHE, ws + OFF_WHHE, ws + OFF_WIHD,
      ws + OFF_WHHD, ws + OFF_WRED, ws + OFF_WOUT,
      (float*)d_out);
}
